// PC_13632226197689
// MI455X (gfx1250) — matrix operand order repaired and checked by reading, not compiled, not run on hardware
//
#include <hip/hip_runtime.h>
#include <math.h>

// ---------------------------------------------------------------------------
// Predictive-coding network on MI455X (gfx1250).
//   * All GEMMs: bf16 WMMA (v_wmma_f32_16x16x32_bf16), fp32 accumulate.
//   * Weights converted fp32->bf16 once (159 MB < 192 MB L2) and kept
//     L2-resident: all large fp32 streams (p/e/be) use non-temporal hints so
//     they do not evict the weights.
//   * Global->LDS tile staging uses gfx1250 async loads
//     (global_load_async_to_lds_b128, ASYNCcnt) with s_wait_asynccnt fences.
//   * 8 waves/block, 128x256 C tile, 64x64 per wave => 16 WMMA per K-step per
//     wave at 1.0 ds_load_b128 per WMMA.
// ---------------------------------------------------------------------------

typedef __bf16 bf16_t;
typedef __attribute__((ext_vector_type(16))) __bf16 v16bf;
typedef __attribute__((ext_vector_type(8)))  __bf16 v8bf;
typedef __attribute__((ext_vector_type(8)))  float  v8f;

union FragBF { v16bf v; v8bf h[2]; };

#define BM 128
#define BN 256
#define BK 32
#define PK 40   // padded LDS row (bf16 elems) to stagger bank accesses

// C[M,N] (fp32, row-major) = A[M,K] (bf16, row-major) @ W[N,K]^T (bf16, row-major)
__global__ __launch_bounds__(256) void k_gemm_bf16(
    const bf16_t* __restrict__ A, const bf16_t* __restrict__ W,
    float* __restrict__ C, int M, int N, int K)
{
    __shared__ bf16_t sA[2][BM][PK];   // 20.0 KB
    __shared__ bf16_t sB[2][BN][PK];   // 40.0 KB

    const int tid  = threadIdx.x;
    const int lane = tid & 31;
    const int wave = tid >> 5;        // 8 waves
    const int wm   = wave >> 2;       // 0..1  (64-row strip)
    const int wn   = wave & 3;        // 0..3  (64-col strip)
    const int m0   = blockIdx.y * BM;
    const int n0   = blockIdx.x * BN;
    const int half = lane >> 4;       // lane half selects K sub-range
    const int l16  = lane & 15;

    v8f acc[4][4];
    const v8f vzero = {0.f,0.f,0.f,0.f,0.f,0.f,0.f,0.f};
#pragma unroll
    for (int i = 0; i < 4; ++i)
#pragma unroll
        for (int j = 0; j < 4; ++j)
            acc[i][j] = vzero;

    // LDS byte offsets: flat addr low 32 bits == addrspace(3) offset.
    const unsigned sA0 = (unsigned)(uintptr_t)(void*)&sA[0][0][0];
    const unsigned sB0 = (unsigned)(uintptr_t)(void*)&sB[0][0][0];

    // Per-thread 16-byte chunk: row r0 (+64*i), cols c0..c0+7.
    const int r0 = tid >> 2;
    const int c0 = (tid & 3) << 3;

    // Async global->LDS staging (ASYNCcnt). A: 2 chunks, B: 4 chunks / thread.
    auto loadTileAsync = [&](int buf, int k0) {
        const bf16_t* gA = A + (size_t)(m0 + r0) * K + (size_t)(k0 + c0);
        const bf16_t* gB = W + (size_t)(n0 + r0) * K + (size_t)(k0 + c0);
        const unsigned lA = sA0 + (unsigned)buf * (BM * PK * 2)
                                + (unsigned)(r0 * PK * 2 + c0 * 2);
        const unsigned lB = sB0 + (unsigned)buf * (BN * PK * 2)
                                + (unsigned)(r0 * PK * 2 + c0 * 2);
#pragma unroll
        for (int i = 0; i < 2; ++i) {
            unsigned long long ga = (unsigned long long)(uintptr_t)(gA + (size_t)i * 64 * K);
            unsigned la = lA + (unsigned)(i * 64 * PK * 2);
            asm volatile("global_load_async_to_lds_b128 %0, %1, off"
                         :: "v"(la), "v"(ga) : "memory");
        }
#pragma unroll
        for (int i = 0; i < 4; ++i) {
            unsigned long long gb = (unsigned long long)(uintptr_t)(gB + (size_t)i * 64 * K);
            unsigned lb = lB + (unsigned)(i * 64 * PK * 2);
            asm volatile("global_load_async_to_lds_b128 %0, %1, off"
                         :: "v"(lb), "v"(gb) : "memory");
        }
    };

    loadTileAsync(0, 0);
    asm volatile("s_wait_asynccnt 0x0" ::: "memory");
    __syncthreads();

    const int nK = K / BK;
    for (int kt = 0; kt < nK; ++kt) {
        const int buf = kt & 1;
        const int k0  = kt * BK;
        if (kt + 1 < nK) loadTileAsync(buf ^ 1, k0 + BK);
        if (kt + 2 < nK) {  // global_prefetch_b8 hint two tiles ahead
            __builtin_prefetch(A + (size_t)(m0 + r0) * K + (size_t)(k0 + 2 * BK + c0), 0, 1);
            __builtin_prefetch(W + (size_t)(n0 + r0) * K + (size_t)(k0 + 2 * BK + c0), 0, 1);
        }

        // A fragment (16x32): lane m holds row m; halves carry
        // K = {base..base+7, base+16..base+23}, base = half*8.
        v16bf aF[4];
#pragma unroll
        for (int mi = 0; mi < 4; ++mi) {
            const bf16_t* p = &sA[buf][wm * 64 + mi * 16 + l16][half * 8];
            FragBF f;
            f.h[0] = *(const v8bf*)(p);
            f.h[1] = *(const v8bf*)(p + 16);
            aF[mi] = f.v;
        }
        // B fragment (32x16): lane n holds W row n; K = {base..base+7, base+16..base+23}, base = half*8.
        v16bf bF[4];
#pragma unroll
        for (int ni = 0; ni < 4; ++ni) {
            const bf16_t* p = &sB[buf][wn * 64 + ni * 16 + l16][half * 8];
            FragBF f;
            f.h[0] = *(const v8bf*)(p);
            f.h[1] = *(const v8bf*)(p + 16);
            bF[ni] = f.v;
        }

#pragma unroll
        for (int mi = 0; mi < 4; ++mi)
#pragma unroll
            for (int ni = 0; ni < 4; ++ni)
                acc[mi][ni] = __builtin_amdgcn_wmma_f32_16x16x32_bf16(
                    false, aF[mi], false, bF[ni], (short)0, acc[mi][ni], false, false);

        if (kt + 1 < nK) asm volatile("s_wait_asynccnt 0x0" ::: "memory");
        __syncthreads();
    }

    // C/D layout: VGPR r, lanes 0-15 -> (M=r, N=lane); lanes 16-31 -> (M=8+r).
    // Non-temporal stores: keep the fp32 result stream from evicting weights.
#pragma unroll
    for (int mi = 0; mi < 4; ++mi) {
#pragma unroll
        for (int ni = 0; ni < 4; ++ni) {
            const int col     = n0 + wn * 64 + ni * 16 + l16;
            const int rowBase = m0 + wm * 64 + mi * 16 + half * 8;
#pragma unroll
            for (int r = 0; r < 8; ++r)
                __builtin_nontemporal_store(acc[mi][ni][r],
                                            &C[(size_t)(rowBase + r) * N + col]);
        }
    }
}

// ---------------------------- elementwise kernels ---------------------------

__device__ __forceinline__ float sigm(float x) { return 1.0f / (1.0f + __expf(-x)); }

__global__ void k_cvt_bf16(const float* __restrict__ s, bf16_t* __restrict__ d, int n) {
    for (int i = blockIdx.x * blockDim.x + threadIdx.x; i < n; i += gridDim.x * blockDim.x)
        d[i] = (bf16_t)s[i];
}

// a = relu(tanh(p)); write fp32 + bf16 copies
__global__ void k_act(const float* __restrict__ p, float* __restrict__ af,
                      bf16_t* __restrict__ ab, int n) {
    for (int i = blockIdx.x * blockDim.x + threadIdx.x; i < n; i += gridDim.x * blockDim.x) {
        float t = tanhf(__builtin_nontemporal_load(&p[i]));
        float v = t > 0.0f ? t : 0.0f;
        af[i] = v;
        ab[i] = (bf16_t)v;
    }
}

// e = a_next - pred(p); pred = sigmoid for top layer else tanh
__global__ void k_err(const float* __restrict__ an, const float* __restrict__ p,
                      float* __restrict__ e, int n, int top) {
    for (int i = blockIdx.x * blockDim.x + threadIdx.x; i < n; i += gridDim.x * blockDim.x) {
        float pv = __builtin_nontemporal_load(&p[i]);
        float pr = top ? sigm(pv) : tanhf(pv);
        __builtin_nontemporal_store(an[i] - pr, &e[i]);
    }
}

// delta = bf16(e * act'(p))
__global__ void k_delta(const float* __restrict__ e, const float* __restrict__ p,
                        bf16_t* __restrict__ d, int n, int top) {
    for (int i = blockIdx.x * blockDim.x + threadIdx.x; i < n; i += gridDim.x * blockDim.x) {
        float pv = __builtin_nontemporal_load(&p[i]);
        float fd;
        if (top) { float s = sigm(pv); fd = s * (1.0f - s); }
        else     { float t = tanhf(pv); fd = 1.0f - t * t; }
        d[i] = (bf16_t)(__builtin_nontemporal_load(&e[i]) * fd);
    }
}

// a = relu(a + 0.1*be - 0.1*e_prev); write fp32 + bf16 copies
__global__ void k_upd(float* __restrict__ af, bf16_t* __restrict__ ab,
                      const float* __restrict__ be, const float* __restrict__ ep, int n) {
    for (int i = blockIdx.x * blockDim.x + threadIdx.x; i < n; i += gridDim.x * blockDim.x) {
        float v = af[i] + 0.1f * __builtin_nontemporal_load(&be[i])
                        - 0.1f * __builtin_nontemporal_load(&ep[i]);
        v = v > 0.0f ? v : 0.0f;
        af[i] = v;
        ab[i] = (bf16_t)v;
    }
}

__global__ void k_sig(const float* __restrict__ p, float* __restrict__ o, int n) {
    for (int i = blockIdx.x * blockDim.x + threadIdx.x; i < n; i += gridDim.x * blockDim.x)
        __builtin_nontemporal_store(sigm(__builtin_nontemporal_load(&p[i])), &o[i]);
}

// ------------------------------- orchestration ------------------------------

static inline void gemm(const bf16_t* A, const bf16_t* W, float* C,
                        int M, int N, int K, hipStream_t s) {
    dim3 g((unsigned)(N / BN), (unsigned)(M / BM));
    k_gemm_bf16<<<g, 256, 0, s>>>(A, W, C, M, N, K);
}

extern "C" void kernel_launch(void* const* d_in, const int* in_sizes, int n_in,
                              void* d_out, int out_size, void* d_ws, size_t ws_size,
                              hipStream_t stream) {
    (void)in_sizes; (void)n_in; (void)out_size; (void)ws_size;

    const float* x      = (const float*)d_in[0];
    const float* target = (const float*)d_in[1];
    const float* Wf[4]  = {(const float*)d_in[2], (const float*)d_in[3],
                           (const float*)d_in[4], (const float*)d_in[5]};
    const float* We[3]  = {(const float*)d_in[6], (const float*)d_in[7],
                           (const float*)d_in[8]};

    const int B = 2048;
    const int L[5] = {1024, 4096, 4096, 4096, 1024};

    // Carve workspace (~585 MB; bf16 weights+acts, fp32 p/e/scratch).
    char* w = (char*)d_ws;
    auto carve = [&](size_t elems, size_t esz) -> void* {
        void* p = (void*)w;
        w += (elems * esz + 255) & ~(size_t)255;
        return p;
    };

    bf16_t* Wf_bf[4]; size_t wf_sz[4];
    for (int i = 0; i < 4; ++i) {
        wf_sz[i] = (size_t)L[i + 1] * (size_t)L[i];
        Wf_bf[i] = (bf16_t*)carve(wf_sz[i], 2);
    }
    bf16_t* We_bf[3]; size_t we_sz[3];
    for (int i = 0; i < 3; ++i) {
        we_sz[i] = (size_t)L[i + 1] * (size_t)L[i + 2];
        We_bf[i] = (bf16_t*)carve(we_sz[i], 2);
    }
    bf16_t* a_bf[4];
    for (int i = 0; i < 4; ++i) a_bf[i] = (bf16_t*)carve((size_t)B * L[i], 2);
    float* a_f[4] = {nullptr, nullptr, nullptr, nullptr};
    for (int i = 1; i < 4; ++i) a_f[i] = (float*)carve((size_t)B * L[i], 4);
    float* p_[4]; float* e_[4];
    for (int i = 0; i < 4; ++i) p_[i] = (float*)carve((size_t)B * L[i + 1], 4);
    for (int i = 0; i < 4; ++i) e_[i] = (float*)carve((size_t)B * L[i + 1], 4);
    bf16_t* delta = (bf16_t*)carve((size_t)B * 4096, 2);
    float*  be    = (float*)carve((size_t)B * 4096, 4);

    const int EB = 256;
    auto ewg = [](size_t n) -> unsigned {
        unsigned g = (unsigned)((n + 1023) / 1024);
        return g ? g : 1u;
    };

    // --- one-time fp32 -> bf16 conversions (weights then live in L2) ---
    for (int i = 0; i < 4; ++i)
        k_cvt_bf16<<<ewg(wf_sz[i]), EB, 0, stream>>>(Wf[i], Wf_bf[i], (int)wf_sz[i]);
    for (int i = 0; i < 3; ++i)
        k_cvt_bf16<<<ewg(we_sz[i]), EB, 0, stream>>>(We[i], We_bf[i], (int)we_sz[i]);
    k_cvt_bf16<<<ewg((size_t)B * L[0]), EB, 0, stream>>>(x, a_bf[0], B * L[0]);

    // --- initial feedforward pass; top activation clamped to target ---
    gemm(a_bf[0], Wf_bf[0], p_[0], B, L[1], L[0], stream);
    k_act<<<ewg((size_t)B * L[1]), EB, 0, stream>>>(p_[0], a_f[1], a_bf[1], B * L[1]);
    gemm(a_bf[1], Wf_bf[1], p_[1], B, L[2], L[1], stream);
    k_act<<<ewg((size_t)B * L[2]), EB, 0, stream>>>(p_[1], a_f[2], a_bf[2], B * L[2]);
    gemm(a_bf[2], Wf_bf[2], p_[2], B, L[3], L[2], stream);
    k_act<<<ewg((size_t)B * L[3]), EB, 0, stream>>>(p_[2], a_f[3], a_bf[3], B * L[3]);
    gemm(a_bf[3], Wf_bf[3], p_[3], B, L[4], L[3], stream);

    // --- 5 relaxation iterations ---
    for (int it = 0; it < 5; ++it) {
        // errors from the state at iteration start
        k_err<<<ewg((size_t)B * L[1]), EB, 0, stream>>>(a_f[1], p_[0], e_[0], B * L[1], 0);
        k_err<<<ewg((size_t)B * L[2]), EB, 0, stream>>>(a_f[2], p_[1], e_[1], B * L[2], 0);
        k_err<<<ewg((size_t)B * L[3]), EB, 0, stream>>>(a_f[3], p_[2], e_[2], B * L[3], 0);
        k_err<<<ewg((size_t)B * L[4]), EB, 0, stream>>>(target, p_[3], e_[3], B * L[4], 1);

        // l = 1
        k_delta<<<ewg((size_t)B * L[2]), EB, 0, stream>>>(e_[1], p_[1], delta, B * L[2], 0);
        gemm(delta, We_bf[0], be, B, L[1], L[2], stream);
        k_upd<<<ewg((size_t)B * L[1]), EB, 0, stream>>>(a_f[1], a_bf[1], be, e_[0], B * L[1]);
        // l = 2
        k_delta<<<ewg((size_t)B * L[3]), EB, 0, stream>>>(e_[2], p_[2], delta, B * L[3], 0);
        gemm(delta, We_bf[1], be, B, L[2], L[3], stream);
        k_upd<<<ewg((size_t)B * L[2]), EB, 0, stream>>>(a_f[2], a_bf[2], be, e_[1], B * L[2]);
        // l = 3 (top uses sigmoid derivative)
        k_delta<<<ewg((size_t)B * L[4]), EB, 0, stream>>>(e_[3], p_[3], delta, B * L[4], 1);
        gemm(delta, We_bf[2], be, B, L[3], L[4], stream);
        k_upd<<<ewg((size_t)B * L[3]), EB, 0, stream>>>(a_f[3], a_bf[3], be, e_[2], B * L[3]);

        // recompute p1..p3 (p0 is invariant: a0 = x never changes)
        gemm(a_bf[1], Wf_bf[1], p_[1], B, L[2], L[1], stream);
        gemm(a_bf[2], Wf_bf[2], p_[2], B, L[3], L[2], stream);
        gemm(a_bf[3], Wf_bf[3], p_[3], B, L[4], L[3], stream);
    }

    // final output: sigmoid of the top pre-activation
    k_sig<<<ewg((size_t)B * L[4]), EB, 0, stream>>>(p_[3], (float*)d_out, B * L[4]);
}
